// DinoV2Loss_21191368638714
// MI455X (gfx1250) — compile-verified, matrix operand order checked by reading, not run on hardware
//
#include <hip/hip_runtime.h>
#include <hip/hip_bf16.h>
#include <math.h>

// DETR-style loss for B=128, Q=2048, C=365, T=50.
// Roofline: dominated by one 383MB read of pred_logits -> ~16.5us @ 23.3TB/s.
// CE kernel: wave owns 16 rows; sum(exp) done on the matrix unit via
// v_wmma_f32_16x16x32_f16 with B=ones (hi/lo f16 split for ~f32 accuracy),
// co-executing with VALU v_exp_f32. Deterministic: no float atomics.
// Tail masking is done with address-clamp + value-select (no EXEC branches).

typedef __attribute__((ext_vector_type(16))) _Float16 v16h;
typedef __attribute__((ext_vector_type(8)))  float    v8f;

#define Bn 128
#define Qn 2048
#define Cn 365
#define Tn 50
#define NCHUNK 12              // 12 * 32 = 384 padded classes
#define TILES_PER_BLOCK 8     // 8 waves/block, one 16-row tile per wave
#define TILES_TOTAL ((Bn * Qn) / 16)   // 16384
#define TILES_PER_BATCH (Qn / 16)      // 128

// ---------------------------------------------------------------------------
// Kernel 1: per-batch L1 matching + bbox L1 partials + label scatter.
// One block per batch. Target boxes staged to LDS with async-to-LDS (gfx1250).
// ---------------------------------------------------------------------------
__global__ __launch_bounds__(256) void k_match(
    const float* __restrict__ pred_boxes,     // [B,Q,4]
    const float* __restrict__ target_boxes,   // [B,T,4]
    const int*   __restrict__ target_labels,  // [B,T]
    int*   __restrict__ class_targets,        // [B,Q] (pre-zeroed)
    float* __restrict__ bbox_part)            // [B,T] sum|matched - tgt|
{
  __shared__ float sTB[Tn * 4];
  __shared__ int   sClosest[Tn];
  const int b    = blockIdx.x;
  const int tid  = threadIdx.x;
  const int lane = tid & 31;
  const int w    = tid >> 5;

  // Async copy of this batch's 200 target-box floats into LDS.
  if (tid < Tn * 4) {
    unsigned lds_off = (unsigned)(size_t)&sTB[tid];
    const float* gp  = target_boxes + (size_t)b * Tn * 4 + tid;
    asm volatile("global_load_async_to_lds_b32 %0, %1, off"
                 :: "v"(lds_off), "v"(gp) : "memory");
  }
  asm volatile("s_wait_asynccnt 0" ::: "memory");
  __syncthreads();

  // Wave per target: scan all Q pred boxes, argmin of L1 distance.
  for (int t = w; t < Tn; t += 8) {
    const float tx = sTB[t*4+0], ty = sTB[t*4+1];
    const float tz = sTB[t*4+2], tw = sTB[t*4+3];
    float best = 3.0e38f; int bq = 0;
    for (int q = lane; q < Qn; q += 32) {
      const float4 p = *(const float4*)(pred_boxes + ((size_t)b*Qn + q)*4);
      float d = fabsf(p.x-tx) + fabsf(p.y-ty) + fabsf(p.z-tz) + fabsf(p.w-tw);
      if (d < best) { best = d; bq = q; }   // strict < : first occurrence wins
    }
    #pragma unroll
    for (int off = 16; off > 0; off >>= 1) {
      float ob = __shfl_xor(best, off, 32);
      int   oq = __shfl_xor(bq,   off, 32);
      if (ob < best || (ob == best && oq < bq)) { best = ob; bq = oq; }
    }
    if (lane == 0) {
      sClosest[t] = bq;
      const float4 p = *(const float4*)(pred_boxes + ((size_t)b*Qn + bq)*4);
      bbox_part[b*Tn + t] =
          fabsf(p.x-tx) + fabsf(p.y-ty) + fabsf(p.z-tz) + fabsf(p.w-tw);
    }
  }
  __syncthreads();

  // Serial scatter (last-writer-wins on duplicate matches, XLA-like).
  if (tid == 0) {
    for (int t = 0; t < Tn; ++t)
      class_targets[(size_t)b*Qn + sClosest[t]] = target_labels[b*Tn + t];
  }
}

// ---------------------------------------------------------------------------
// Kernel 2: weighted cross-entropy. Wave = one 16-row tile.
// Pass1: per-row max (data lands in L0). Pass2 (cache-hit): exp(x-max),
// hi/lo f16 split, row-sum via WMMA with all-ones B.
// A-fragment layout (16-bit A 16x32): lane L covers row L&15; lanes<16 hold
// K {0..7,16..23}, lanes>=16 hold K {8..15,24..31} of each 32-class chunk.
// Out-of-range classes (365..383) use clamped addresses + v_cndmask selects
// so control flow stays uniform (EXEC all ones, required for WMMA anyway).
// ---------------------------------------------------------------------------
__global__ __launch_bounds__(256) void k_ce(
    const float* __restrict__ logits,         // [B*Q, C]
    const int*   __restrict__ class_targets,  // [B*Q]
    float* __restrict__ ce_num,               // [TILES_TOTAL] sum wt*nll
    float* __restrict__ ce_den)               // [TILES_TOTAL] sum wt
{
  __shared__ float sSum[TILES_PER_BLOCK][16];
  const int tid  = threadIdx.x;
  const int lane = tid & 31;
  const int w    = tid >> 5;
  const int tile = blockIdx.x * TILES_PER_BLOCK + w;
  const int rowbase = tile * 16;
  const int r = lane & 15;
  const int h = (lane >> 4) * 8;   // 0 for lanes 0-15, 8 for lanes 16-31
  const float* rowp = logits + (size_t)(rowbase + r) * Cn;

  // ---- pass 1: row max over this lane's elements (clamped loads) ----
  float m = -3.0e38f;
  for (int k = 0; k < NCHUNK; ++k) {
    const int base = 32*k + h;
    #pragma unroll
    for (int g = 0; g < 2; ++g) {
      const int cb = base + g*16;
      #pragma unroll
      for (int j = 0; j < 8; ++j) {
        const int c  = cb + j;
        const int cc = (c < Cn) ? c : 0;         // clamp: load stays in-bounds
        const float v = rowp[cc];
        m = fmaxf(m, (c < Cn) ? v : -3.0e38f);   // select, not branch
      }
    }
  }
  m = fmaxf(m, __shfl_xor(m, 16, 32));   // both half-row lanes -> full row max

  // ---- pass 2: exp + WMMA row-sum (every D column = row sum, B = ones) ----
  v16h ones;
  #pragma unroll
  for (int i = 0; i < 16; ++i) ones[i] = (_Float16)1.0f;
  v8f acc = {0.f, 0.f, 0.f, 0.f, 0.f, 0.f, 0.f, 0.f};

  for (int k = 0; k < NCHUNK; ++k) {
    const int base = 32*k + h;
    v16h ahi, alo;
    #pragma unroll
    for (int g = 0; g < 2; ++g) {
      const int cb = base + g*16;
      #pragma unroll
      for (int j = 0; j < 8; ++j) {
        const int c  = cb + j;
        const int cc = (c < Cn) ? c : 0;
        const float v = rowp[cc];                // in-bounds; exp(v-m) <= 1
        float e = __expf(v - m);
        e = (c < Cn) ? e : 0.0f;                 // mask padding via select
        const _Float16 hi = (_Float16)e;
        ahi[g*8 + j] = hi;
        alo[g*8 + j] = (_Float16)(e - (float)hi);   // residual, ~22-bit total
      }
    }
    acc = __builtin_amdgcn_wmma_f32_16x16x32_f16(
        false, ahi, false, ones, (short)0, acc, false, false);
    acc = __builtin_amdgcn_wmma_f32_16x16x32_f16(
        false, alo, false, ones, (short)0, acc, false, false);
  }

  // D layout: VGPR i = row i (lanes 0-15) / row 8+i (lanes 16-31), all N equal.
  if (lane == 0) {
    #pragma unroll
    for (int i = 0; i < 8; ++i) sSum[w][i] = acc[i];
  }
  if (lane == 16) {
    #pragma unroll
    for (int i = 0; i < 8; ++i) sSum[w][8 + i] = acc[i];
  }
  __syncthreads();

  // Per-row NLL: lane r (0..15) finishes row rowbase+r.
  float pn = 0.f, pd = 0.f;
  if (lane < 16) {
    const int   rowG = rowbase + lane;
    const int   t    = class_targets[rowG];
    const float lt   = logits[(size_t)rowG * Cn + t];  // cache hit
    const float s    = sSum[w][lane];
    const float nll  = -(lt - m - __logf(s));
    const float wt   = (t == 0) ? 0.1f : 1.0f;
    pn = wt * nll; pd = wt;
  }
  #pragma unroll
  for (int off = 16; off > 0; off >>= 1) {
    pn += __shfl_xor(pn, off, 32);
    pd += __shfl_xor(pd, off, 32);
  }
  if (lane == 0) { ce_num[tile] = pn; ce_den[tile] = pd; }
}

// ---------------------------------------------------------------------------
// Kernel 3: deterministic finalize. Thread b reduces its batch serially.
// ---------------------------------------------------------------------------
__global__ __launch_bounds__(128) void k_final(
    const float* __restrict__ ce_num, const float* __restrict__ ce_den,
    const float* __restrict__ bbox_part, float* __restrict__ out)
{
  __shared__ float red[Bn];
  const int b = threadIdx.x;
  float num = 0.f, den = 0.f;
  for (int i = 0; i < TILES_PER_BATCH; ++i) {
    num += ce_num[b*TILES_PER_BATCH + i];
    den += ce_den[b*TILES_PER_BATCH + i];
  }
  float bb = 0.f;
  for (int t = 0; t < Tn; ++t) bb += bbox_part[b*Tn + t];
  bb *= (1.0f / (float)(Tn * 4));
  red[b] = 2.0f * (num / den) + 5.0f * bb;
  __syncthreads();
  #pragma unroll
  for (int off = Bn/2; off > 0; off >>= 1) {
    if (b < off) red[b] += red[b + off];
    __syncthreads();
  }
  if (b == 0) out[0] = red[0] * (1.0f / (float)Bn);
}

// ---------------------------------------------------------------------------
extern "C" void kernel_launch(void* const* d_in, const int* in_sizes, int n_in,
                              void* d_out, int out_size, void* d_ws, size_t ws_size,
                              hipStream_t stream) {
  const float* pred_logits   = (const float*)d_in[0];
  const float* pred_boxes    = (const float*)d_in[1];
  const float* target_boxes  = (const float*)d_in[2];
  const int*   target_labels = (const int*)d_in[3];
  float* out = (float*)d_out;

  // Workspace layout (~1.2 MB): class_targets, ce partials, bbox partials.
  char* ws = (char*)d_ws;
  int*   class_targets = (int*)ws;                                   // B*Q ints
  float* ce_num    = (float*)(ws + (size_t)Bn * Qn * sizeof(int));   // 16384
  float* ce_den    = ce_num + TILES_TOTAL;                           // 16384
  float* bbox_part = ce_den + TILES_TOTAL;                           // 6400

  hipMemsetAsync(class_targets, 0, (size_t)Bn * Qn * sizeof(int), stream);
  k_match<<<Bn, 256, 0, stream>>>(pred_boxes, target_boxes, target_labels,
                                  class_targets, bbox_part);
  k_ce<<<TILES_TOTAL / TILES_PER_BLOCK, 256, 0, stream>>>(
      pred_logits, class_targets, ce_num, ce_den);
  k_final<<<1, Bn, 0, stream>>>(ce_num, ce_den, bbox_part, out);
}
